// CGNNOFDM_87694642249960
// MI455X (gfx1250) — matrix operand order repaired and checked by reading, not compiled, not run on hardware
//
#include <hip/hip_runtime.h>
#include <hip/hip_bf16.h>

typedef __attribute__((ext_vector_type(8)))  _Float16 v8h;
typedef __attribute__((ext_vector_type(16))) _Float16 v16h;
typedef __attribute__((ext_vector_type(8)))  float    v8f;
typedef __attribute__((ext_vector_type(4)))  float    v4f;

#define NB   16
#define NTXc 4
#define FFc  512
#define TTc  14
#define FTc  (FFc*TTc)          // 7168
#define NPc  (NB*NTXc*FTc)      // 458752 positions, multiple of 1024

// ---------------------------------------------------------------------------
// Build z0 = concat(y_rep, pe_rep) -> f16 [NP, 64] (34 real channels, pad 0)
// ---------------------------------------------------------------------------
__global__ __launch_bounds__(256) void k_build_z0(const float* __restrict__ y,
                                                  const float* __restrict__ pe,
                                                  _Float16* __restrict__ z0) {
  int idx = blockIdx.x * 256 + threadIdx.x;   // NP*64 threads
  int c   = idx & 63;
  int pos = idx >> 6;
  int ft  = pos % FTc;
  int img = pos / FTc;
  int b   = img >> 2;
  int tx  = img & 3;
  float v = 0.f;
  if (c < 32)      v = y[(b * FTc + ft) * 32 + c];
  else if (c < 34) v = pe[(tx * FTc + ft) * 2 + (c - 32)];
  z0[idx] = (_Float16)v;
}

// ---------------------------------------------------------------------------
// 3x3 depthwise conv over (F,T) per image, SAME padding, f16 in/out.
// Thread per (pos, c); consecutive threads -> consecutive channels (coalesced).
// Channels >= Creal are written as zero (K-padding for the following WMMA GEMM).
// ---------------------------------------------------------------------------
__global__ __launch_bounds__(256) void k_dw3x3(const _Float16* __restrict__ in,
                                               const float* __restrict__ w,
                                               _Float16* __restrict__ out,
                                               int Creal, int Cs) {
  int idx = blockIdx.x * 256 + threadIdx.x;   // NP*Cs threads
  int c   = idx % Cs;
  int pos = idx / Cs;
  if (c >= Creal) { out[idx] = (_Float16)0.f; return; }
  int t    = pos % TTc;
  int f    = (pos / TTc) % FFc;
  int base = pos - (f * TTc + t);             // start of this image
  float acc = 0.f;
#pragma unroll
  for (int ky = 0; ky < 3; ++ky) {
    int ff = f + ky - 1;
    if (ff < 0 || ff >= FFc) continue;
#pragma unroll
    for (int kx = 0; kx < 3; ++kx) {
      int tt = t + kx - 1;
      if (tt < 0 || tt >= TTc) continue;
      acc += (float)in[(base + ff * TTc + tt) * Cs + c] * w[(ky * 3 + kx) * Creal + c];
    }
  }
  out[idx] = (_Float16)acc;
}

// ---------------------------------------------------------------------------
// Leave-one-out aggregation over TX + build zc = concat(a, s, pe) f16 [NP, 96]
// (66 real channels, pad 0)
// ---------------------------------------------------------------------------
__global__ __launch_bounds__(256) void k_agg_concat(const _Float16* __restrict__ sp,
                                                    const _Float16* __restrict__ s,
                                                    const float* __restrict__ pe,
                                                    const float* __restrict__ act,
                                                    _Float16* __restrict__ zc) {
  int idx  = blockIdx.x * 256 + threadIdx.x;  // NB*FT*32 threads
  int c    = idx & 31;
  int rest = idx >> 5;
  int ft   = rest % FTc;
  int b    = rest / FTc;
  float v[NTXc], m[NTXc];
  float sum = 0.f, cnt = 0.f;
#pragma unroll
  for (int tx = 0; tx < NTXc; ++tx) {
    m[tx] = act[b * NTXc + tx];
    float val = (float)sp[((b * NTXc + tx) * FTc + ft) * 32 + c] * m[tx];
    v[tx] = val; sum += val; cnt += m[tx];
  }
  float p = cnt - 1.f; if (p < 0.f) p = 0.f;
  float inv = (p == 0.f) ? 1.f : 1.f / fmaxf(p, 1e-10f);
#pragma unroll
  for (int tx = 0; tx < NTXc; ++tx) {
    int pos = (b * NTXc + tx) * FTc + ft;
    zc[pos * 96 + c]      = (_Float16)((sum - v[tx]) * inv);
    zc[pos * 96 + 32 + c] = s[pos * 32 + c];
    if (c < 2) zc[pos * 96 + 64 + c] = (_Float16)pe[(tx * FTc + ft) * 2 + c];
    else       zc[pos * 96 + 64 + c] = (_Float16)0.f;   // pad 66..95
  }
}

// ---------------------------------------------------------------------------
// WMMA GEMM: out[M,NN] = A[M,K] * W[K,NN] (+bias)(+resid)(relu), f16 A, f32 W.
// 256 thr = 8 wave32. Each wave: preload all B-fragments (W^T, from LDS) into
// registers once, then loop over MT=8 M-tiles of 16 rows: clause of b128
// A-loads -> back-to-back v_wmma_f32_16x16x32_f16 -> bias/relu -> LDS-staged
// fully-coalesced b128 store (output tile contiguous, ldo == NN).
// ---------------------------------------------------------------------------
template <int KPAD, int NN, bool RELU, bool BIAS, bool RESID, bool F32OUT>
__global__ __launch_bounds__(256) void k_gemm(const _Float16* __restrict__ A, int lda,
                                              const float* __restrict__ W, int Kreal,
                                              const float* __restrict__ bias,
                                              const _Float16* __restrict__ resid,
                                              _Float16* __restrict__ outH,
                                              float* __restrict__ outF) {
  constexpr int MT = 8;                              // M tiles per wave
  __shared__ __align__(16) _Float16 wT[64 * 96];     // [n][k], stride 96 halves
  __shared__ __align__(16) _Float16 stage[8 * 1024]; // 2KB per wave (f16/f32 view)
  __shared__ float bS[64];
  for (int i = threadIdx.x; i < 64 * 96; i += 256) wT[i] = (_Float16)0.f;
  if (threadIdx.x < 64)
    bS[threadIdx.x] = (BIAS && threadIdx.x < NN) ? bias[threadIdx.x] : 0.f;
  __syncthreads();
  for (int i = threadIdx.x; i < Kreal * NN; i += 256) {
    int k = i / NN, n = i % NN;
    wT[n * 96 + k] = (_Float16)W[i];
  }
  __syncthreads();

  const int wv   = threadIdx.x >> 5;
  const int lane = threadIdx.x & 31;
  const int hf   = lane >> 4;         // 0/1 half of wave
  const int ln   = lane & 15;
  const int koffA = hf ? 8 : 0;

  constexpr int NK = KPAD / 32;
  constexpr int NT = (NN + 15) / 16;

  // ---- preload all B fragments into registers (loop-invariant) ----
  v16h bfrag[NT][NK];
#pragma unroll
  for (int nt = 0; nt < NT; ++nt) {
    const _Float16* wp = &wT[(nt * 16 + ln) * 96];
#pragma unroll
    for (int kc = 0; kc < NK; ++kc) {
      int kb = kc * 32 + (hf ? 8 : 0);
      v8h blo = *(const v8h*)(wp + kb);
      v8h bhi = *(const v8h*)(wp + kb + 16);
      bfrag[nt][kc] =
          __builtin_shufflevector(blo, bhi, 0,1,2,3,4,5,6,7,8,9,10,11,12,13,14,15);
    }
  }

  _Float16* st  = &stage[wv * 1024];  // wave-private 2KB
  float*    stF = (float*)st;
  const int tile0 = (blockIdx.x * 8 + wv) * MT;

#pragma unroll 2
  for (int mt = 0; mt < MT; ++mt) {
    const int mbase = (tile0 + mt) * 16;
    const int row   = mbase + ln;

    v16h afrag[NK];
#pragma unroll
    for (int kc = 0; kc < NK; ++kc) {
      const _Float16* ap = A + (long)row * lda + kc * 32;
      v8h lo = *(const v8h*)(ap + koffA);
      v8h hi = *(const v8h*)(ap + 16 + koffA);
      afrag[kc] =
          __builtin_shufflevector(lo, hi, 0,1,2,3,4,5,6,7,8,9,10,11,12,13,14,15);
    }

#pragma unroll
    for (int nt = 0; nt < NT; ++nt) {
      v8f acc = {};
#pragma unroll
      for (int kc = 0; kc < NK; ++kc)
        acc = __builtin_amdgcn_wmma_f32_16x16x32_f16(false, afrag[kc], false,
                                                     bfrag[nt][kc], (short)0, acc,
                                                     false, false);
      float bn  = bS[nt * 16 + ln];
      int   col = nt * 16 + ln;
#pragma unroll
      for (int vv = 0; vv < 8; ++vv) {
        int mloc = vv + 8 * hf;         // 0..15 row within tile
        float val = acc[vv];
        if (BIAS) val += bn;
        if (RELU) val = fmaxf(val, 0.f);
        if (col < NN) {
          if (F32OUT) stF[mloc * NN + col] = val;
          else        st [mloc * NN + col] = (_Float16)val;
        }
      }
    }

    // ---- stream the contiguous 16xNN tile out of LDS, fully coalesced ----
    if (F32OUT) {
      float* op = outF + (long)mbase * NN;
      if constexpr (NN % 8 == 0) {        // NN=32: 4 iters of b128
        constexpr int IT = (16 * NN) / (32 * 4);
#pragma unroll
        for (int it = 0; it < IT; ++it) {
          int off = (it * 32 + lane) * 4;
          *(v4f*)(op + off) = *(const v4f*)(stF + off);
        }
      } else {                            // NN=4: 2 iters of b32
        constexpr int IT = (16 * NN) / 32;
#pragma unroll
        for (int it = 0; it < IT; ++it) {
          int off = it * 32 + lane;
          op[off] = stF[off];
        }
      }
    } else {
      _Float16* op = outH + (long)mbase * NN;
      constexpr int IT = NN / 16;         // 2 (NN=32) or 4 (NN=64) iters of b128
#pragma unroll
      for (int it = 0; it < IT; ++it) {
        int off = it * 256 + lane * 8;
        v8h v = *(const v8h*)(st + off);
        if (RESID) {
          v8h r = *(const v8h*)(resid + (long)mbase * NN + off);
#pragma unroll
          for (int j = 0; j < 8; ++j) v[j] = (_Float16)((float)v[j] + (float)r[j]);
        }
        *(v8h*)(op + off) = v;
      }
    }
  }
}

// ---------------------------------------------------------------------------
extern "C" void kernel_launch(void* const* d_in, const int* in_sizes, int n_in,
                              void* d_out, int out_size, void* d_ws, size_t ws_size,
                              hipStream_t stream) {
  const float* y   = (const float*)d_in[0];
  const float* pe  = (const float*)d_in[1];
  const float* act = (const float*)d_in[2];
  auto P = [&](int i) { return (const float*)d_in[i]; };

  char* ws = (char*)d_ws;
  _Float16* A96 = (_Float16*)(ws);                 // NP*96 f16 (88.1 MB)
  _Float16* B96 = (_Float16*)(ws + 88080384ull);   // NP*96 f16
  _Float16* C64 = (_Float16*)(ws + 176160768ull);  // NP*64 f16 (58.7 MB)
  _Float16* S32 = (_Float16*)(ws + 234881024ull);  // NP*32 f16 (29.4 MB)
  _Float16* Q32 = (_Float16*)(ws + 264241152ull);  // NP*32 f16

  float* outLLR = (float*)d_out;                   // (B,NTX,F,T,4)
  float* outCH  = outLLR + (long)NPc * 4;          // (B,NTX,F,T,32)

  const dim3 blk(256);
  const int gM   = NPc / (128 * 8);     // 448 GEMM blocks (8 M-tiles per wave)
  const int g64  = NPc * 64 / 256;      // 114688
  const int g96  = NPc * 96 / 256;      // 172032
  const int gAgg = NB * FTc * 32 / 256; // 14336

  // ---- init conv stack: relu(pw(dw(z0))) then pw(dw(.)) -> s ----
  k_build_z0<<<g64, blk, 0, stream>>>(y, pe, A96);
  k_dw3x3<<<g64, blk, 0, stream>>>(A96, P(3), B96, 34, 64);
  k_gemm<64, 64, true,  false, false, false><<<gM, blk, 0, stream>>>(
      B96, 64, P(4), 34, nullptr, nullptr, A96, nullptr);
  k_dw3x3<<<g64, blk, 0, stream>>>(A96, P(5), B96, 64, 64);
  k_gemm<64, 32, false, false, false, false><<<gM, blk, 0, stream>>>(
      B96, 64, P(6), 64, nullptr, nullptr, S32, nullptr);

  // ---- 2 message-passing iterations ----
  for (int it = 0; it < 2; ++it) {
    int b0 = 7 + it * 8;
    // agg MLP: 32->64 relu+bias, 64->32 +bias
    k_gemm<32, 64, true,  true,  false, false><<<gM, blk, 0, stream>>>(
        S32, 32, P(b0 + 0), 32, P(b0 + 1), nullptr, C64, nullptr);
    k_gemm<64, 32, false, true,  false, false><<<gM, blk, 0, stream>>>(
        C64, 64, P(b0 + 2), 64, P(b0 + 3), nullptr, Q32, nullptr);
    // leave-one-out aggregation + concat(a, s, pe) -> zc [NP,96]
    k_agg_concat<<<gAgg, blk, 0, stream>>>(Q32, S32, pe, act, A96);
    // upd conv stack: relu(pw66->64(dw66)) then pw64->32(dw64) + residual s
    k_dw3x3<<<g96, blk, 0, stream>>>(A96, P(b0 + 4), B96, 66, 96);
    k_gemm<96, 64, true,  false, false, false><<<gM, blk, 0, stream>>>(
        B96, 96, P(b0 + 5), 66, nullptr, nullptr, C64, nullptr);
    k_dw3x3<<<g64, blk, 0, stream>>>(C64, P(b0 + 6), B96, 64, 64);
    k_gemm<64, 32, false, false, true,  false><<<gM, blk, 0, stream>>>(
        B96, 64, P(b0 + 7), 64, nullptr, S32, S32, nullptr);
  }

  // ---- readouts ----
  // llr: 32->64 relu+bias, 64->4 +bias -> f32 d_out
  k_gemm<32, 64, true,  true,  false, false><<<gM, blk, 0, stream>>>(
      S32, 32, P(23), 32, P(24), nullptr, C64, nullptr);
  k_gemm<64, 4,  false, true,  false, true><<<gM, blk, 0, stream>>>(
      C64, 64, P(25), 64, P(26), nullptr, nullptr, outLLR);
  // h_hat: 32->64 relu+bias, 64->32 +bias -> f32 d_out (offset)
  k_gemm<32, 64, true,  true,  false, false><<<gM, blk, 0, stream>>>(
      S32, 32, P(27), 32, P(28), nullptr, C64, nullptr);
  k_gemm<64, 32, false, true,  false, true><<<gM, blk, 0, stream>>>(
      C64, 64, P(29), 64, P(30), nullptr, nullptr, outCH);
}
